// GCNModel_64295660421451
// MI455X (gfx1250) — compile-verified, matrix operand order checked by reading, not run on hardware
//
#include <hip/hip_runtime.h>
#include <hip/hip_fp16.h>

typedef __attribute__((ext_vector_type(16))) _Float16 v16h;
typedef __attribute__((ext_vector_type(8)))  float    v8f;

union Frag16 {
    v16h v;
    _Float16 h[16];
    unsigned u[8];
};

// K offset of packed pair p (elements 2p,2p+1) per ISA 16-bit A/B layout:
// lanes 0-15 hold K {0..7, 16..23}, lanes 16-31 hold K {8..15, 24..31}.
__device__ __forceinline__ int pair_koff(int lane, int p) {
    return ((lane >> 4) << 3) + (p < 4 ? 2 * p : 2 * p + 8);
}

// --- async-to-LDS pointer types (signature confirmed by round-2 diagnostic:
//     param0 = int __vector(4) AS1*, param1 = AS3 counterpart) --------------
typedef int v4i __attribute__((vector_size(16)));
typedef __attribute__((address_space(1))) v4i as1_v4i;
typedef __attribute__((address_space(3))) v4i as3_v4i;

__device__ __forceinline__ as1_v4i* to_global_v4(const void* p) {
    return (as1_v4i*)(unsigned long long)(uintptr_t)p;
}
__device__ __forceinline__ as3_v4i* to_lds_v4(const void* p) {
    return (as3_v4i*)(unsigned)(uintptr_t)p;
}

#if __has_builtin(__builtin_amdgcn_global_load_async_to_lds_b128)
#define HAVE_ASYNC_LDS 1
#endif

#if __has_builtin(__builtin_amdgcn_tensor_load_to_lds) && \
    __has_builtin(__builtin_amdgcn_s_wait_tensorcnt)
#define HAVE_TDM 1
#if __clang_major__ >= 23
#define HAVE_TDM_6ARG 1
#endif
#endif

#ifdef HAVE_TDM
typedef unsigned tdm_u32x4 __attribute__((ext_vector_type(4)));
typedef int      tdm_i32x8 __attribute__((ext_vector_type(8)));
typedef int      tdm_i32x4 __attribute__((ext_vector_type(4)));

// 2-D f16 tile load via Tensor Data Mover. Tile = tile_rows x 64 elements,
// row stride 64 elements in memory; LDS rows padded by 16B per 128B row
// (pad_interval=4 -> every 32 DWORDs, pad_amount=3 -> 4 DWORDs) to give the
// 72-half LDS stride. Rows >= tensor_rows read as zero (OOB zero-fill).
__device__ __forceinline__ void tdm_load_tile_f16(const void* gsrc, unsigned lds_off,
                                                  unsigned tensor_rows,
                                                  unsigned tile_rows) {
    const unsigned long long ga = (unsigned long long)(uintptr_t)gsrc;
    tdm_u32x4 g0;
    g0.x = 1u;                                              // count=1 (valid)
    g0.y = lds_off;                                         // lds_addr (bytes)
    g0.z = (unsigned)ga;                                    // global_addr[31:0]
    g0.w = (unsigned)((ga >> 32) & 0x01FFFFFFull) | (2u << 30);  // [56:32]+type=2
    tdm_i32x8 g1;
    g1[0] = (int)((1u << 16) | (1u << 20) | (4u << 22) | (3u << 25));
    //            data_size=2B  pad_en      pad_int=32dw  pad_amt=4dw
    g1[1] = (int)(64u << 16);                               // tensor_dim0 = 64
    g1[2] = (int)((tensor_rows & 0xFFFFu) << 16);           // tensor_dim1 lo16
    g1[3] = (int)(((tensor_rows >> 16) & 0xFFFFu) | (64u << 16)); // hi16, tile_dim0=64
    g1[4] = (int)(tile_rows & 0xFFFFu);                     // tile_dim1
    g1[5] = 64;                                             // tensor_dim0_stride
    g1[6] = 0;
    g1[7] = 0;
    const tdm_i32x4 z4 = {0, 0, 0, 0};
#ifdef HAVE_TDM_6ARG
    const tdm_i32x8 z8 = {0, 0, 0, 0, 0, 0, 0, 0};
    __builtin_amdgcn_tensor_load_to_lds(g0, g1, z4, z4, z8, 0);
#else
    __builtin_amdgcn_tensor_load_to_lds(g0, g1, z4, z4, 0);
#endif
}
#endif  // HAVE_TDM

// ---------------------------------------------------------------------------
// Generic WMMA GEMM: C[n x dout] = A[n x din] @ W[din x dout], f32 in/out,
// f16 multiply, f32 accumulate. One 16x16 tile per wave.
// ---------------------------------------------------------------------------
__global__ __launch_bounds__(256) void gemm_wmma_f32(
    float* __restrict__ C, const float* __restrict__ A,
    const float* __restrict__ W, int n, int din, int dout) {
    const int lane = threadIdx.x & 31;
    const int wave = threadIdx.x >> 5;
    const int colTiles = dout >> 4;
    const int tiles = (n >> 4) * colTiles;
    const int tile = blockIdx.x * 8 + wave;
    if (tile >= tiles) return;                  // wave-uniform branch
    const int rowBase = (tile / colTiles) << 4;
    const int colBase = (tile % colTiles) << 4;
    const int m = lane & 15;
    const int kgrp = (lane >> 4) << 3;

    v8f acc = {};
    for (int k0 = 0; k0 < din; k0 += 32) {
        if (k0 + 32 < din) {                    // prefetch next K-step (WGP scope)
            __builtin_prefetch(&A[(size_t)(rowBase + m) * din + k0 + 32], 0, 3);
            __builtin_prefetch(&W[(size_t)(k0 + 32) * dout + colBase + m], 0, 3);
        }
        Frag16 a, b;
#pragma unroll
        for (int e = 0; e < 16; ++e) {
            const int k = k0 + kgrp + (e < 8 ? e : e + 8);
            a.h[e] = (_Float16)A[(size_t)(rowBase + m) * din + k];
            b.h[e] = (_Float16)W[(size_t)k * dout + colBase + m];
        }
        acc = __builtin_amdgcn_wmma_f32_16x16x32_f16(
            false, a.v, false, b.v, (short)0, acc, false, false);
    }

    const int mrow = rowBase + ((lane >> 4) << 3);
#pragma unroll
    for (int r = 0; r < 8; ++r)
        C[(size_t)(mrow + r) * dout + colBase + m] = acc[r];
}

// ---------------------------------------------------------------------------
// Final decoder: out[i,j] = sigmoid(sum_k g[i,k] * h[j,k]), K = 64, f16 in.
// Block tile 128(i) x 64(j); operands staged to LDS via TDM (preferred),
// async global->LDS B128, or plain loads. Padded 72-half stride throughout.
// ---------------------------------------------------------------------------
#define BG_I 128
#define BG_J 64
#define LSTR 72

__global__ __launch_bounds__(256) void biggemm_sigmoid(
    float* __restrict__ out, const __half* __restrict__ gq,
    const __half* __restrict__ hq, int n) {
    const _Float16* __restrict__ g = reinterpret_cast<const _Float16*>(gq);
    const _Float16* __restrict__ h = reinterpret_cast<const _Float16*>(hq);
    __shared__ _Float16 sG[BG_I * LSTR];
    __shared__ _Float16 sH[BG_J * LSTR];

    const int iBase = blockIdx.y * BG_I;
    const int jBase = blockIdx.x * BG_J;
    const int tid = threadIdx.x;

#ifdef HAVE_TDM
    // One TDM descriptor per operand tile, issued by wave 0 (EXEC-independent,
    // wave-level op); OOB rows zero-filled by tensor_dim1, LDS pad by D#.
    if (tid == 0) {
        tdm_load_tile_f16(g + (size_t)iBase * 64, (unsigned)(uintptr_t)sG,
                          (unsigned)(n - iBase), BG_I);
        tdm_load_tile_f16(h + (size_t)jBase * 64, (unsigned)(uintptr_t)sH,
                          (unsigned)(n - jBase), BG_J);
    }
    if (tid < 32) __builtin_amdgcn_s_wait_tensorcnt(0);
    __syncthreads();
#else
    // Stage G tile: 128 rows x 64 halves in 16-byte chunks.
    for (int c = tid; c < (BG_I * 64) / 8; c += 256) {
        const int row = c >> 3, cc = (c & 7) << 3;
        const int gi = iBase + row;
#ifdef HAVE_ASYNC_LDS
        if (gi < n)
            __builtin_amdgcn_global_load_async_to_lds_b128(
                to_global_v4(g + (size_t)gi * 64 + cc),
                to_lds_v4(sG + row * LSTR + cc), 0, 0);
        else
            *(uint4*)(sG + row * LSTR + cc) = make_uint4(0u, 0u, 0u, 0u);
#else
        uint4 v = make_uint4(0u, 0u, 0u, 0u);
        if (gi < n) v = *(const uint4*)(g + (size_t)gi * 64 + cc);
        *(uint4*)(sG + row * LSTR + cc) = v;
#endif
    }
    // Stage H tile: 64 rows x 64 halves.
    for (int c = tid; c < (BG_J * 64) / 8; c += 256) {
        const int row = c >> 3, cc = (c & 7) << 3;
        const int gj = jBase + row;
#ifdef HAVE_ASYNC_LDS
        if (gj < n)
            __builtin_amdgcn_global_load_async_to_lds_b128(
                to_global_v4(h + (size_t)gj * 64 + cc),
                to_lds_v4(sH + row * LSTR + cc), 0, 0);
        else
            *(uint4*)(sH + row * LSTR + cc) = make_uint4(0u, 0u, 0u, 0u);
#endif
#ifndef HAVE_ASYNC_LDS
        uint4 v = make_uint4(0u, 0u, 0u, 0u);
        if (gj < n) v = *(const uint4*)(h + (size_t)gj * 64 + cc);
        *(uint4*)(sH + row * LSTR + cc) = v;
#endif
    }
#ifdef HAVE_ASYNC_LDS
#if __has_builtin(__builtin_amdgcn_s_wait_asynccnt)
    __builtin_amdgcn_s_wait_asynccnt(0);
#else
    asm volatile("s_wait_asynccnt 0x0" ::: "memory");
#endif
#endif
    __syncthreads();
#endif  // HAVE_TDM

    const int lane = tid & 31, wave = tid >> 5;
    const int m = lane & 15;
    const int aRow = wave * 16 + m;

    // A fragments depend only on (wave, K): hoist out of the column-tile loop.
    Frag16 a0, a1;
#pragma unroll
    for (int p = 0; p < 8; ++p) {
        const int ko = pair_koff(lane, p);
        a0.u[p] = *(const unsigned*)(sG + aRow * LSTR + ko);
        a1.u[p] = *(const unsigned*)(sG + aRow * LSTR + 32 + ko);
    }

    for (int ct = 0; ct < 4; ++ct) {
        Frag16 b0, b1;
#pragma unroll
        for (int p = 0; p < 8; ++p) {
            const int ko = pair_koff(lane, p);
            b0.u[p] = *(const unsigned*)(sH + (ct * 16 + m) * LSTR + ko);
            b1.u[p] = *(const unsigned*)(sH + (ct * 16 + m) * LSTR + 32 + ko);
        }
        v8f acc = {};
        acc = __builtin_amdgcn_wmma_f32_16x16x32_f16(
            false, a0.v, false, b0.v, (short)0, acc, false, false);
        acc = __builtin_amdgcn_wmma_f32_16x16x32_f16(
            false, a1.v, false, b1.v, (short)0, acc, false, false);

        const int gi0 = iBase + wave * 16 + ((lane >> 4) << 3);
        const int gj = jBase + ct * 16 + m;
        if (gj < n) {
#pragma unroll
            for (int r = 0; r < 8; ++r) {
                const int gi = gi0 + r;
                if (gi < n)
                    out[(size_t)gi * n + gj] = 1.0f / (1.0f + __expf(-acc[r]));
            }
        }
    }
}

// ---------------------------------------------------------------------------
// SPMM: acc[row[e]] += val[e] * t[col[e]], 4 features per thread, f32 atomics.
// ---------------------------------------------------------------------------
__global__ void spmm_atomic(float* __restrict__ acc, const float* __restrict__ t,
                            const int* __restrict__ er, const int* __restrict__ ec,
                            const float* __restrict__ ev, int E, int d, int dlog) {
    const long long idx = (long long)blockIdx.x * blockDim.x + threadIdx.x;
    const long long e = idx >> dlog;
    const int q = (int)(idx & ((1 << dlog) - 1));
    if (e >= E) return;
    const int r = er[e], c = ec[e];
    const float v = ev[e];
    const float4 x = *(const float4*)(t + (size_t)c * d + q * 4);
    float* dst = acc + (size_t)r * d + q * 4;
    atomicAdd(dst + 0, v * x.x);   // global_atomic_add_f32
    atomicAdd(dst + 1, v * x.y);
    atomicAdd(dst + 2, v * x.z);
    atomicAdd(dst + 3, v * x.w);
}

__global__ void zero_f32(float* __restrict__ a, long long cnt) {
    const long long i = (long long)blockIdx.x * blockDim.x + threadIdx.x;
    if (i < cnt) a[i] = 0.0f;
}

__global__ void elu_inplace(float* __restrict__ a, long long cnt) {
    const long long i = (long long)blockIdx.x * blockDim.x + threadIdx.x;
    if (i < cnt) {
        const float v = a[i];
        a[i] = v > 0.0f ? v : (__expf(v) - 1.0f);
    }
}

__global__ void cvt_f32_f16(__half* __restrict__ o, const float* __restrict__ a,
                            long long cnt) {
    const long long i = (long long)blockIdx.x * blockDim.x + threadIdx.x;
    if (i < cnt) o[i] = __float2half(a[i]);
}

__global__ void symmetrize64(float* __restrict__ S, const float* __restrict__ Wb) {
    const int idx = blockIdx.x * blockDim.x + threadIdx.x;
    if (idx < 64 * 64) {
        const int i = idx >> 6, j = idx & 63;
        S[idx] = 0.5f * (Wb[i * 64 + j] + Wb[j * 64 + i]);
    }
}

// ---------------------------------------------------------------------------
extern "C" void kernel_launch(void* const* d_in, const int* in_sizes, int n_in,
                              void* d_out, int out_size, void* d_ws, size_t ws_size,
                              hipStream_t stream) {
    const float* x  = (const float*)d_in[0];
    const int*   er = (const int*)d_in[1];
    const int*   ec = (const int*)d_in[2];
    const float* ev = (const float*)d_in[3];
    const float* W0 = (const float*)d_in[4];
    const float* W1 = (const float*)d_in[5];
    const float* W2 = (const float*)d_in[6];
    const float* Wb = (const float*)d_in[7];
    const int N = in_sizes[0] / 128;
    const int E = in_sizes[1];
    float* out = (float*)d_out;

    // Workspace layout (~12.8 MB)
    float* bufA = (float*)d_ws;                    // N*128 f32
    float* bufB = bufA + (size_t)N * 128;          // N*128 f32
    float* Ssym = bufB + (size_t)N * 128;          // 64*64 f32
    __half* gh  = (__half*)(Ssym + 64 * 64);       // N*64 f16 (g = h3 @ S)
    __half* hh  = gh + (size_t)N * 64;             // N*64 f16 (h3)

    auto gemm = [&](float* C, const float* A, const float* W, int din, int dout) {
        const int tiles = (N >> 4) * (dout >> 4);
        gemm_wmma_f32<<<(tiles + 7) / 8, 256, 0, stream>>>(C, A, W, N, din, dout);
    };
    auto ew = [&](long long cnt) { return (unsigned)((cnt + 255) / 256); };

    // Layer 1: h1 = elu(spmm(x @ W0))
    gemm(bufA, x, W0, 128, 128);
    zero_f32<<<ew((long long)N * 128), 256, 0, stream>>>(bufB, (long long)N * 128);
    spmm_atomic<<<(unsigned)(((long long)E * 32 + 255) / 256), 256, 0, stream>>>(
        bufB, bufA, er, ec, ev, E, 128, 5);
    elu_inplace<<<ew((long long)N * 128), 256, 0, stream>>>(bufB, (long long)N * 128);

    // Layer 2: h2 = elu(spmm(h1 @ W1))
    gemm(bufA, bufB, W1, 128, 128);
    zero_f32<<<ew((long long)N * 128), 256, 0, stream>>>(bufB, (long long)N * 128);
    spmm_atomic<<<(unsigned)(((long long)E * 32 + 255) / 256), 256, 0, stream>>>(
        bufB, bufA, er, ec, ev, E, 128, 5);
    elu_inplace<<<ew((long long)N * 128), 256, 0, stream>>>(bufB, (long long)N * 128);

    // Layer 3: h3 = elu(spmm(h2 @ W2)), d -> 64
    gemm(bufA, bufB, W2, 128, 64);
    zero_f32<<<ew((long long)N * 64), 256, 0, stream>>>(bufB, (long long)N * 64);
    spmm_atomic<<<(unsigned)(((long long)E * 16 + 255) / 256), 256, 0, stream>>>(
        bufB, bufA, er, ec, ev, E, 64, 4);
    elu_inplace<<<ew((long long)N * 64), 256, 0, stream>>>(bufB, (long long)N * 64);

    // Decoder: S = (Wb + Wb^T)/2 ; g = h3 @ S ; out = sigmoid(g @ h3^T)
    symmetrize64<<<16, 256, 0, stream>>>(Ssym, Wb);
    gemm(bufA, bufB, Ssym, 64, 64);
    cvt_f32_f16<<<ew((long long)N * 64), 256, 0, stream>>>(gh, bufA, (long long)N * 64);
    cvt_f32_f16<<<ew((long long)N * 64), 256, 0, stream>>>(hh, bufB, (long long)N * 64);

    dim3 grid((N + BG_J - 1) / BG_J, (N + BG_I - 1) / BG_I);
    biggemm_sigmoid<<<grid, 256, 0, stream>>>(out, gh, hh, N);
}
